// ScaledDotProductAttention_16045997818194
// MI455X (gfx1250) — matrix operand order repaired and checked by reading, not compiled, not run on hardware
//
#include <hip/hip_runtime.h>

#define B_ 16
#define S_ 4096
#define D_ 64

typedef __attribute__((ext_vector_type(16))) _Float16 v16h;
typedef __attribute__((ext_vector_type(8)))  _Float16 v8h;
typedef __attribute__((ext_vector_type(4)))  _Float16 v4h;
typedef __attribute__((ext_vector_type(8)))  float    v8f;

__device__ __forceinline__ v8f wmma16(v16h a, v16h b, v8f c) {
    // D = A(16x32 f16) * B(32x16 f16) + C(16x16 f32)
    return __builtin_amdgcn_wmma_f32_16x16x32_f16(
        false, a, false, b, (short)0, c, false, false);
}

// B fragment: 16 contiguous halves (16B-aligned); caller applies the lane-half base.
__device__ __forceinline__ v16h ldB16(const _Float16* p) {
    v8h lo = *(const v8h*)p;
    v8h hi = *(const v8h*)(p + 8);
    return __builtin_shufflevector(lo, hi, 0,1,2,3,4,5,6,7,8,9,10,11,12,13,14,15);
}

// A fragment: lane<16 holds K {0..15}, lane>=16 holds K {16..31}
__device__ __forceinline__ v16h ldA16(const _Float16* rowp, int lane) {
    const _Float16* p = rowp + ((lane & 16) ? 16 : 0);
    v8h lo = *(const v8h*)p;
    v8h hi = *(const v8h*)(p + 8);
    return __builtin_shufflevector(lo, hi, 0,1,2,3,4,5,6,7,8,9,10,11,12,13,14,15);
}

// ---- CDNA5 async global->LDS copy (no VGPR round trip, tracked by ASYNCcnt) ----
__device__ __forceinline__ void async_b128(const _Float16* g, _Float16* l) {
    // LDS aperture flat address carries the wave-relative LDS offset in addr[31:0]
    unsigned lds = (unsigned)(unsigned long long)l;
    asm volatile("global_load_async_to_lds_b128 %0, %1, off"
                 :: "v"(lds), "v"(g) : "memory");
}

__device__ __forceinline__ void wait_async0() {
#if __has_builtin(__builtin_amdgcn_s_wait_asynccnt)
    __builtin_amdgcn_s_wait_asynccnt(0);
#else
    asm volatile("s_wait_asynccnt 0x0" ::: "memory");
#endif
}

// ---------------------------------------------------------------------------
// Prep kernel: K(f32)->Kh(f16, [b][s][d]) and V(f32)->VTh(f16, [b][d][s]).
// One-time ~50MB pass (~2us at 23.3TB/s); hoists conversion+transpose out of
// the attention hot loop.
// ---------------------------------------------------------------------------
__global__ __launch_bounds__(256)
void prep_kernel(const float* __restrict__ K, const float* __restrict__ V,
                 _Float16* __restrict__ Kh, _Float16* __restrict__ VTh) {
    __shared__ alignas(32) _Float16 T[64 * 72];
    const int tid = threadIdx.x;
    const int b   = blockIdx.x >> 6;
    const int st  = blockIdx.x & 63;
    const size_t inOff = ((size_t)b * S_ + st * 64) * D_;

    const float* Kt = K + inOff;
    _Float16*   Kho = Kh + inOff;
#pragma unroll
    for (int it = 0; it < 4; ++it) {
        int f = tid + it * 256;
        float4 kv = *(const float4*)(Kt + f * 4);
        v4h h = {(_Float16)kv.x, (_Float16)kv.y, (_Float16)kv.z, (_Float16)kv.w};
        *(v4h*)(Kho + f * 4) = h;
    }

    const float* Vt = V + inOff;
#pragma unroll
    for (int it = 0; it < 4; ++it) {
        int f   = tid + it * 256;
        int row = f >> 4;
        int c4  = (f & 15) << 2;
        float4 vv = *(const float4*)(Vt + row * 64 + c4);
        T[(c4 + 0) * 72 + row] = (_Float16)vv.x;
        T[(c4 + 1) * 72 + row] = (_Float16)vv.y;
        T[(c4 + 2) * 72 + row] = (_Float16)vv.z;
        T[(c4 + 3) * 72 + row] = (_Float16)vv.w;
    }
    __syncthreads();

    _Float16* Vo = VTh + (size_t)b * D_ * S_ + st * 64;
    const int r0 = tid >> 3, c8 = (tid & 7) << 3;
#pragma unroll
    for (int it = 0; it < 2; ++it) {
        int r = r0 + it * 32;
        v8h x = *(const v8h*)&T[r * 72 + c8];
        *(v8h*)(Vo + (size_t)r * S_ + c8) = x;
    }
}

// ---------------------------------------------------------------------------
// Flash attention main kernel: transposed-GEMM formulation, f16 WMMA,
// double-buffered async global->LDS staging (one barrier per tile).
// ---------------------------------------------------------------------------
__global__ __launch_bounds__(256)
void fattn_kernel(const float* __restrict__ Q, const _Float16* __restrict__ Kh,
                  const _Float16* __restrict__ VTh, float* __restrict__ O) {
    __shared__ alignas(32) _Float16 Kl2[2][64 * 72];   // K tile  [key][d], double buffered
    __shared__ alignas(32) _Float16 VT2[2][64 * 72];   // V tile  [d][key], double buffered
    __shared__ alignas(32) _Float16 Pst[8 * 16 * 72];  // per-wave P^T stage [m][key]

    const int tid  = threadIdx.x;
    const int lane = tid & 31;
    const int w    = tid >> 5;
    const int col  = lane & 15;       // query row m within wave tile
    const int hhi  = lane >> 4;

    const int b     = blockIdx.x >> 5;
    const int qBase = (blockIdx.x & 31) * 128;

    const size_t    bOff = (size_t)b * S_ * D_;
    const _Float16* Kg   = Kh  + bOff;                  // [s][d]
    const _Float16* VTg  = VTh + (size_t)b * D_ * S_;   // [d][s]

    // Q as B-fragments of Q^T; fold 1/sqrt(D)*log2(e) so exps are bare v_exp_f32
    const float SC = 0.125f * 1.44269504088896f;
    const float* Qrow = Q + bOff + (size_t)(qBase + w * 16 + col) * D_;
    const int dB = (lane & 16) ? 16 : 0;
    v16h qb0, qb1;
#pragma unroll
    for (int j = 0; j < 16; ++j) {
        qb0[j] = (_Float16)(Qrow[dB + j]      * SC);
        qb1[j] = (_Float16)(Qrow[32 + dB + j] * SC);
    }

    v16h onesA;
#pragma unroll
    for (int j = 0; j < 16; ++j) onesA[j] = (_Float16)1.0f;

    v8f oc[4] = {};
    float mstat = -1e30f, lstat = 0.f;
    _Float16* Pw = &Pst[w * (16 * 72)];

    const int r0 = tid >> 3, c8 = (tid & 7) << 3;

    // prologue: DMA tile 0 into buffer 0 (4 async b128 per thread)
#pragma unroll
    for (int it = 0; it < 2; ++it) {
        int r = r0 + it * 32;
        async_b128(Kg  + r * 64 + c8,            &Kl2[0][r * 72 + c8]);
        async_b128(VTg + (size_t)r * S_ + c8,    &VT2[0][r * 72 + c8]);
    }

    for (int kt = 0; kt < S_ / 64; ++kt) {
        wait_async0();        // this wave's copies for tile kt have landed
        __syncthreads();      // all waves' copies visible; old buffer fully consumed

        if (kt + 1 < S_ / 64) {   // DMA next tile into the other buffer, overlaps compute
            const _Float16* Ksn = Kg  + (kt + 1) * 64 * D_;
            const _Float16* Vsn = VTg + (kt + 1) * 64;
            _Float16* KlN = Kl2[(kt + 1) & 1];
            _Float16* VTN = VT2[(kt + 1) & 1];
#pragma unroll
            for (int it = 0; it < 2; ++it) {
                int r = r0 + it * 32;
                async_b128(Ksn + r * 64 + c8,         &KlN[r * 72 + c8]);
                async_b128(Vsn + (size_t)r * S_ + c8, &VTN[r * 72 + c8]);
            }
        }

        const _Float16* KlB = Kl2[kt & 1];
        const _Float16* VTB = VT2[kt & 1];

        // ---- GEMM1: S^T = K * Q^T (C rows = keys, cols = m) ----
        v8f sc[4];
#pragma unroll
        for (int cb = 0; cb < 4; ++cb) {
            const _Float16* krow = &KlB[(cb * 16 + col) * 72];
            v16h ka0 = ldA16(krow, lane);
            v16h ka1 = ldA16(krow + 32, lane);
            v8f acc = {};
            acc = wmma16(ka0, qb0, acc);
            acc = wmma16(ka1, qb1, acc);
            sc[cb] = acc;   // log2-domain scores, keys cb*16 + r + 8*hhi
        }

        // ---- softmax over keys: in-lane tree + one cross-half exchange ----
        v8f mv;
#pragma unroll
        for (int r = 0; r < 8; ++r)
            mv[r] = fmaxf(fmaxf(sc[0][r], sc[1][r]), fmaxf(sc[2][r], sc[3][r]));
        float mloc = fmaxf(fmaxf(fmaxf(mv[0], mv[1]), fmaxf(mv[2], mv[3])),
                           fmaxf(fmaxf(mv[4], mv[5]), fmaxf(mv[6], mv[7])));
        float mr   = fmaxf(mloc, __shfl_xor(mloc, 16));
        float mnew = fmaxf(mstat, mr);
        float corr = __builtin_amdgcn_exp2f(mstat - mnew);
        mstat = mnew;
        lstat *= corr;
#pragma unroll
        for (int nb = 0; nb < 4; ++nb) oc[nb] *= corr;

        // ---- P^T = exp2(S^T - m), packed key-contiguous stores ----
#pragma unroll
        for (int cb = 0; cb < 4; ++cb) {
            v8h ph;
#pragma unroll
            for (int r = 0; r < 8; ++r)
                ph[r] = (_Float16)__builtin_amdgcn_exp2f(sc[cb][r] - mnew);
            *(v8h*)&Pw[col * 72 + cb * 16 + 8 * hhi] = ph;
        }
        __builtin_amdgcn_wave_barrier();

        const _Float16* prow = &Pw[col * 72] + ((lane & 16) ? 16 : 0);
        v16h pb0 = ldB16(prow);        // keys 0..31
        v16h pb1 = ldB16(prow + 32);   // keys 32..63

        // ---- row-sum of P via WMMA vs ones ----
        v8f lac = {};
        lac = wmma16(onesA, pb0, lac);
        lac = wmma16(onesA, pb1, lac);
        lstat += lac[0];

        // ---- GEMM2: O^T += V^T * P^T ----
#pragma unroll
        for (int nb = 0; nb < 4; ++nb) {
            const _Float16* vrow = &VTB[(nb * 16 + col) * 72];
            v16h va0 = ldA16(vrow, lane);
            v16h va1 = ldA16(vrow + 32, lane);
            oc[nb] = wmma16(va0, pb0, oc[nb]);
            oc[nb] = wmma16(va1, pb1, oc[nb]);
        }
    }

    // ---- normalize + store: contiguous float4 runs in O[m][d] ----
    const float inv = 1.0f / lstat;
    float* Orow = O + bOff + (size_t)(qBase + w * 16 + col) * D_;
#pragma unroll
    for (int nb = 0; nb < 4; ++nb) {
        float4 f0 = {oc[nb][0] * inv, oc[nb][1] * inv, oc[nb][2] * inv, oc[nb][3] * inv};
        float4 f1 = {oc[nb][4] * inv, oc[nb][5] * inv, oc[nb][6] * inv, oc[nb][7] * inv};
        *(float4*)(Orow + nb * 16 + 8 * hhi)     = f0;
        *(float4*)(Orow + nb * 16 + 8 * hhi + 4) = f1;
    }
}

extern "C" void kernel_launch(void* const* d_in, const int* in_sizes, int n_in,
                              void* d_out, int out_size, void* d_ws, size_t ws_size,
                              hipStream_t stream) {
    (void)in_sizes; (void)n_in; (void)out_size; (void)ws_size;
    const float* Q = (const float*)d_in[0];
    const float* K = (const float*)d_in[1];
    const float* V = (const float*)d_in[2];
    float* O = (float*)d_out;

    // workspace layout: Kh (f16) | VTh (f16), 8 MB each
    _Float16* Kh  = (_Float16*)d_ws;
    _Float16* VTh = Kh + (size_t)B_ * S_ * D_;

    prep_kernel<<<dim3(B_ * (S_ / 64)), dim3(256), 0, stream>>>(K, V, Kh, VTh);
    fattn_kernel<<<dim3(B_ * (S_ / 128)), dim3(256), 0, stream>>>(Q, Kh, VTh, O);
}
